// GKTMultiHeadAttention_45818711113765
// MI455X (gfx1250) — matrix operand order repaired and checked by reading, not compiled, not run on hardware
//
#include <hip/hip_runtime.h>

#define VV 6
#define CC 256
#define HW 16384   // 128*128

typedef __attribute__((ext_vector_type(16))) __bf16   v16bf;
typedef __attribute__((ext_vector_type(8)))  float    v8f;
typedef __attribute__((ext_vector_type(4)))  unsigned v4u;
typedef __attribute__((ext_vector_type(8)))  int      v8i;
typedef __attribute__((ext_vector_type(4)))  int      v4i;

__device__ __forceinline__ unsigned short f2bf(float f) {
  union { float f; unsigned u; } c; c.f = f;
  unsigned u = c.u;
  u += 0x7FFFu + ((u >> 16) & 1u);      // round-to-nearest-even
  return (unsigned short)(u >> 16);
}

// ---------------------------------------------------------------------------
// Prep kernel: G = Wq^T * Wk  (bf16),  Wv -> bf16.  65536 threads, trivial.
// ---------------------------------------------------------------------------
__global__ void prep_kernel(const float* __restrict__ Wq,
                            const float* __restrict__ Wk,
                            const float* __restrict__ Wv,
                            unsigned short* __restrict__ Gbf,
                            unsigned short* __restrict__ Wvbf) {
  int idx = blockIdx.x * blockDim.x + threadIdx.x;   // 0..65535
  int a = idx >> 8, b = idx & 255;
  float acc = 0.f;
  for (int o = 0; o < CC; ++o)
    acc = fmaf(Wq[o * CC + a], Wk[o * CC + b], acc);
  Gbf[idx]  = f2bf(acc);
  Wvbf[idx] = f2bf(Wv[idx]);
}

// ---------------------------------------------------------------------------
// Main kernel.  Per wave: a 16-pixel strip.
//   pass1 (per view): keep the 8 B-frags of the k-column in VGPRs; for each
//     row-tile r reduce over K with 8 chained WMMAs into ONE 8-VGPR acc,
//     dot with q immediately, discard.  att_v per pixel.
//   softmax over views; vbar = sum_v w_v * v_v  (built as B-frags)
//   pass2: out row-tile = Wv-rows x vbar, stored right after its K-reduction.
//   r loops unrolled x4 -> 4 independent WMMA chains to cover XDL hazards.
//   A-frags fetched as 2x ds_load_b128 per fragment.
// ---------------------------------------------------------------------------
__global__ __launch_bounds__(256) void attn_kernel(
    const float* __restrict__ q, const float* __restrict__ k,
    const float* __restrict__ vt, float* __restrict__ out,
    const unsigned short* __restrict__ Gbf,
    const unsigned short* __restrict__ Wvbf) {
  extern __shared__ char smem[];   // 256 KB: [0,128K) = G bf16, [128K,256K) = Wv bf16
  const int tid  = threadIdx.x;
  const int lane = tid & 31;
  const int wave = tid >> 5;
  const int half = lane >> 4;
  const int col  = lane & 15;

  // ---- stage G and Wv into LDS via the Tensor Data Mover ----
#if defined(__gfx1250__) && __has_builtin(__builtin_amdgcn_tensor_load_to_lds)
  if (tid == 0) {
    unsigned lds0 = (unsigned)(size_t)&smem[0];
    unsigned long long ga = (unsigned long long)(size_t)Gbf;
    unsigned long long wa = (unsigned long long)(size_t)Wvbf;
    // D# group1: data_size=2B, tensor 256x256, tile 256x256, stride0=256
    v8i g1 = { 0x00010000, 0x01000000, 0x01000000, 0x01000000,
               0x00000100, 256, 0, 0 };
    v4i z4 = { 0, 0, 0, 0 };
    v8i z8 = { 0, 0, 0, 0, 0, 0, 0, 0 };
    v4u g0a = { 1u, lds0, (unsigned)ga,
                (unsigned)((ga >> 32) & 0x1FFFFFFull) | (2u << 30) };
    __builtin_amdgcn_tensor_load_to_lds(g0a, g1, z4, z4, z8, 0);
    v4u g0b = { 1u, lds0 + 131072u, (unsigned)wa,
                (unsigned)((wa >> 32) & 0x1FFFFFFull) | (2u << 30) };
    __builtin_amdgcn_tensor_load_to_lds(g0b, g1, z4, z4, z8, 0);
    __builtin_amdgcn_s_wait_tensorcnt(0);
  }
  __syncthreads();
#else
  for (int i = tid; i < CC * CC; i += 256) {
    ((unsigned short*)smem)[i]            = Gbf[i];
    ((unsigned short*)(smem + 131072))[i] = Wvbf[i];
  }
  __syncthreads();
#endif

  const char* Gb = (const char*)smem;            // G  bf16 [256][256]
  const char* Wb = (const char*)smem + 131072;   // Wv bf16 [256][256]

  const int p0 = blockIdx.x * 128 + wave * 16;
  const int p  = p0 + col;

  float att[VV];

  // ---------------- pass 1: attention logits per view ----------------
#pragma unroll 1
  for (int vi = 0; vi < VV; ++vi) {
    const float* kv = k + (size_t)vi * CC * HW;
    const float* qv = q + (size_t)vi * CC * HW;

    // all 8 B-fragments of this view's k column (64 VGPRs), reused by all r
    v16bf bfv[8];
#pragma unroll
    for (int kk = 0; kk < 8; ++kk) {
      union { v16bf v; unsigned short u[16]; } bt;
#pragma unroll
      for (int j = 0; j < 8; ++j) {
        int c = kk * 32 + half * 16 + 2 * j;
        bt.u[2 * j]     = f2bf(kv[(size_t)c * HW + p]);
        bt.u[2 * j + 1] = f2bf(kv[(size_t)(c + 1) * HW + p]);
      }
      bfv[kk] = bt.v;
    }

    float d = 0.f;
#pragma unroll 4
    for (int r = 0; r < 16; ++r) {
      v8f accf = (v8f){0, 0, 0, 0, 0, 0, 0, 0};
      int arow = r * 16 + col;
#pragma unroll
      for (int kk = 0; kk < 8; ++kk) {
        union { v16bf v; v4u blk[2]; } af;
        int off = arow * 512 + kk * 64 + half * 32;
        af.blk[0] = *(const v4u*)(Gb + off);
        af.blk[1] = *(const v4u*)(Gb + off + 16);
        accf = __builtin_amdgcn_wmma_f32_16x16x32_bf16(
            false, af.v, false, bfv[kk], (short)0, accf, false, false);
      }
      // fold this row-tile of t into the per-pixel dot with q
#pragma unroll
      for (int j = 0; j < 8; ++j) {
        int c = r * 16 + j + half * 8;
        d = fmaf(qv[(size_t)c * HW + p], accf[j], d);
      }
    }
    d += __shfl_xor(d, 16, 32);   // combine the two half-wave partial sums
    att[vi] = d;
  }

  // ---------------- softmax over the 6 views (per pixel) ----------------
  float mx = att[0];
#pragma unroll
  for (int vi = 1; vi < VV; ++vi) mx = fmaxf(mx, att[vi]);
  float s = 0.f;
#pragma unroll
  for (int vi = 0; vi < VV; ++vi) { att[vi] = __expf(att[vi] - mx); s += att[vi]; }
  float inv = 1.f / s;
#pragma unroll
  for (int vi = 0; vi < VV; ++vi) att[vi] *= inv;

  // ---------------- pass 2: out = Wv @ (sum_v w_v * v_v) ----------------
  // build vbar B-fragments (weighted view-sum of v), 64 VGPRs
  v16bf vb[8];
#pragma unroll
  for (int kk = 0; kk < 8; ++kk) {
    union { v16bf v; unsigned short u[16]; } bt;
#pragma unroll
    for (int j = 0; j < 8; ++j) {
#pragma unroll
      for (int ss = 0; ss < 2; ++ss) {
        int c = kk * 32 + half * 16 + 2 * j + ss;
        float a2 = 0.f;
#pragma unroll
        for (int vi = 0; vi < VV; ++vi)
          a2 = fmaf(att[vi], vt[((size_t)vi * CC + c) * HW + p], a2);
        bt.u[2 * j + ss] = f2bf(a2);
      }
    }
    vb[kk] = bt.v;
  }

#pragma unroll 4
  for (int r = 0; r < 16; ++r) {
    v8f accf = (v8f){0, 0, 0, 0, 0, 0, 0, 0};
    int arow = r * 16 + col;
#pragma unroll
    for (int kk = 0; kk < 8; ++kk) {
      union { v16bf v; v4u blk[2]; } af;
      int off = arow * 512 + kk * 64 + half * 32;
      af.blk[0] = *(const v4u*)(Wb + off);
      af.blk[1] = *(const v4u*)(Wb + off + 16);
      accf = __builtin_amdgcn_wmma_f32_16x16x32_bf16(
          false, af.v, false, vb[kk], (short)0, accf, false, false);
    }
    // store this 16-channel row-tile of out
#pragma unroll
    for (int j = 0; j < 8; ++j) {
      int c = r * 16 + j + half * 8;
      out[(size_t)c * HW + p] = accf[j];
    }
  }
}

// ---------------------------------------------------------------------------
extern "C" void kernel_launch(void* const* d_in, const int* in_sizes, int n_in,
                              void* d_out, int out_size, void* d_ws, size_t ws_size,
                              hipStream_t stream) {
  const float* q  = (const float*)d_in[0];
  const float* k  = (const float*)d_in[1];
  const float* v  = (const float*)d_in[2];
  const float* Wq = (const float*)d_in[3];
  const float* Wk = (const float*)d_in[4];
  const float* Wv = (const float*)d_in[5];
  float* out = (float*)d_out;

  unsigned short* Gbf  = (unsigned short*)d_ws;                    // 128 KB
  unsigned short* Wvbf = (unsigned short*)((char*)d_ws + 131072);  // 128 KB

  prep_kernel<<<dim3(256), dim3(256), 0, stream>>>(Wq, Wk, Wv, Gbf, Wvbf);
  attn_kernel<<<dim3(HW / 128), dim3(256), 262144, stream>>>(q, k, v, out, Gbf, Wvbf);
}
